// PKM_23210003268040
// MI455X (gfx1250) — matrix operand order repaired and checked by reading, not compiled, not run on hardware
//
#include <hip/hip_runtime.h>
#include <hip/hip_bf16.h>
#include <math.h>

#define BT     4096
#define DIMX   512
#define DIMQ   1024
#define NHEADS 4
#define NKEYS  256
#define TOPKK  16
#define DHALF  128
#define NEG_INF -3.4e38f

typedef __attribute__((ext_vector_type(16))) __bf16 bf16x16;
typedef __attribute__((ext_vector_type(8)))  float  f32x8;
typedef __attribute__((ext_vector_type(8)))  unsigned short us8;
typedef __attribute__((ext_vector_type(4)))  unsigned u32x4;
typedef __attribute__((ext_vector_type(8)))  unsigned u32x8;

static __device__ __forceinline__ unsigned short f2bf_raw(float f) {
  union { float f; unsigned u; } v; v.f = f;
  unsigned r = (v.u + 0x7FFFu + ((v.u >> 16) & 1u)) >> 16;
  return (unsigned short)r;
}

static __device__ __forceinline__ f32x8 wmma_bf16(bf16x16 a, bf16x16 b, f32x8 c) {
  return __builtin_amdgcn_wmma_f32_16x16x32_bf16(false, a, false, b, (short)0, c,
                                                 false, false);
}

// A fragment 16x32 bf16 (ISA 7.12.2): lane m=lane&15, K runs {kb..kb+7, 16+kb..16+kb+7}
// with kb = (lane>=16)*8. Row-major LDS => two contiguous 16B runs -> ds_load_b128.
static __device__ __forceinline__ bf16x16 load_A_frag(const unsigned short* lds,
                                                      int ldk, int lane) {
  int m  = lane & 15;
  int kb = (lane >> 4) * 8;
  const unsigned short* pp = lds + m * ldk + kb;
  union { us8 h[2]; bf16x16 v; } u;
  u.h[0] = *(const us8*)(pp);
  u.h[1] = *(const us8*)(pp + 16);
  return u.v;
}

// B fragment 32x16 bf16: lane col n=lane&15, K runs {kb..kb+7, 16+kb..16+kb+7} (kb = (lane>=16)*8).
// n-major LDS (k contiguous per key) => two contiguous 16B runs.
static __device__ __forceinline__ bf16x16 load_B_frag(const unsigned short* lds,
                                                      int ldk, int lane) {
  int n  = lane & 15;
  int kb = (lane >> 4) * 8;
  const unsigned short* pp = lds + n * ldk + kb;
  union { us8 h[2]; bf16x16 v; } u;
  u.h[0] = *(const us8*)(pp);
  u.h[1] = *(const us8*)(pp + 16);
  return u.v;
}

// C/D 16x16 f32: VGPR r -> row r (lanes 0-15) or 8+r (lanes 16-31), col = lane&15
static __device__ __forceinline__ void store_C(float* out, int ldn, int lane, f32x8 c) {
  int n  = lane & 15;
  int mb = (lane >> 4) * 8;
#pragma unroll
  for (int r = 0; r < 8; ++r) out[(mb + r) * ldn + n] = c[r];
}

// ---------------------------------------------------------------- GEMM1: Q = X*Wq
#define LDA_G 40   // padded strides: keep 16B alignment, spread banks
#define LDB_G 40
__global__ __launch_bounds__(256) void k_gemm_q(const float* __restrict__ x,
                                                const float* __restrict__ wq,
                                                float* __restrict__ Q) {
  __shared__ unsigned short ldsA[32 * LDA_G];   // row-major 32M x 32K
  __shared__ unsigned short ldsB[256 * LDB_G];  // n-major 256N x 32K
  int tid = threadIdx.x;
  int m0 = blockIdx.x * 32;
  int n0 = blockIdx.y * 256;
  int w = tid >> 5, lane = tid & 31;
  int wm = (w >> 2) * 16;
  int wn = (w & 3) * 64;
  f32x8 acc[4] = {};
  for (int kk = 0; kk < DIMX; kk += 32) {
    __syncthreads();
#pragma unroll
    for (int j = 0; j < 4; ++j) {       // A: 32x32 f32 -> bf16, row-major
      int e = tid + 256 * j;
      int r = e >> 5, c = e & 31;
      ldsA[r * LDA_G + c] = f2bf_raw(x[(size_t)(m0 + r) * DIMX + kk + c]);
    }
#pragma unroll
    for (int j = 0; j < 32; ++j) {      // B: 32k x 256n, stored n-major
      int e = tid + 256 * j;
      int r = e >> 8, c = e & 255;
      ldsB[c * LDB_G + r] = f2bf_raw(wq[(size_t)(kk + r) * DIMQ + n0 + c]);
    }
    __syncthreads();
    bf16x16 a = load_A_frag(ldsA + wm * LDA_G, LDA_G, lane);
#pragma unroll
    for (int t = 0; t < 4; ++t) {
      bf16x16 b = load_B_frag(ldsB + (wn + t * 16) * LDB_G, LDB_G, lane);
      acc[t] = wmma_bf16(a, b, acc[t]);
    }
  }
#pragma unroll
  for (int t = 0; t < 4; ++t)
    store_C(Q + (size_t)(m0 + wm) * DIMQ + n0 + wn + t * 16, DIMQ, lane, acc[t]);
}

// ------------------------------------------------- BatchNorm stats (per column)
__global__ __launch_bounds__(256) void k_bn_stats(const float* __restrict__ Q,
                                                  float* __restrict__ mu,
                                                  float* __restrict__ rstd) {
  __shared__ float s_sum[256], s_sq[256];
  int tid = threadIdx.x;
  int col = blockIdx.x * 32 + (tid & 31);
  int grp = tid >> 5;
  float s = 0.f, s2 = 0.f;
  for (int r = grp; r < BT; r += 8) {
    float v = Q[(size_t)r * DIMQ + col];
    s += v; s2 += v * v;
  }
  s_sum[tid] = s; s_sq[tid] = s2;
  __syncthreads();
  if (tid < 32) {
    float ts = 0.f, t2 = 0.f;
#pragma unroll
    for (int g = 0; g < 8; ++g) { ts += s_sum[g * 32 + tid]; t2 += s_sq[g * 32 + tid]; }
    float m = ts * (1.f / (float)BT);
    float var = t2 * (1.f / (float)BT) - m * m;
    mu[blockIdx.x * 32 + tid]   = m;
    rstd[blockIdx.x * 32 + tid] = rsqrtf(var + 1e-5f);
  }
}

// -------------------------------------------- BatchNorm apply -> bf16 Qn for dots
__global__ __launch_bounds__(256) void k_bn_apply(const float* __restrict__ Q,
                                                  const float* __restrict__ mu,
                                                  const float* __restrict__ rstd,
                                                  const float* __restrict__ gamma,
                                                  const float* __restrict__ beta,
                                                  unsigned short* __restrict__ Qn) {
  size_t base = ((size_t)blockIdx.x * 256 + threadIdx.x) * 4;
#pragma unroll
  for (int j = 0; j < 4; ++j) {
    size_t i = base + j;
    int c = (int)(i & (DIMQ - 1));
    float v = (Q[i] - mu[c]) * rstd[c] * gamma[c] + beta[c];
    Qn[i] = f2bf_raw(v);
  }
}

// ------------------- keys f32 [h][n][p][d] -> bf16 [h][p][n][d] (TDM-friendly)
__global__ __launch_bounds__(256) void k_cvt_keys(const float* __restrict__ keys,
                                                  unsigned short* __restrict__ keysbf) {
  int e0 = (blockIdx.x * 256 + threadIdx.x) * 4;
#pragma unroll
  for (int j = 0; j < 4; ++j) {
    int e = e0 + j;                 // e over h*p*n*d (h:4, p:2, n:256, d:128)
    int d = e & 127;
    int n = (e >> 7) & 255;
    int p = (e >> 15) & 1;
    int h = e >> 16;
    keysbf[e] = f2bf_raw(keys[(((size_t)h * NKEYS + n) * 2 + p) * DHALF + d]);
  }
}

// ---------- dots (WMMA, TDM-staged keys) + per-row top-16, fused in-LDS
#define LDA_D 136   // 128 + 8 pad (TDM pad: 4 dwords every 64 dwords)
__global__ __launch_bounds__(256) void k_dots_topk(const unsigned short* __restrict__ Qn,
                                                   const unsigned short* __restrict__ keysbf,
                                                   float* __restrict__ s1ws,
                                                   int* __restrict__ i1ws) {
  __shared__ unsigned short ldsA[16 * LDA_D];    // 16 rows x 128 d (padded)
  __shared__ unsigned short ldsB[256 * LDA_D];   // 256 keys x 128 d (padded), TDM dest
  __shared__ float scores[16 * 256];
  int tid = threadIdx.x;
  int bt0 = blockIdx.x * 16;
  int hp = blockIdx.y;
  int h = hp >> 1, p = hp & 1;
  int qc0 = p * 512 + h * 128;
  int w = tid >> 5, lane = tid & 31;

  // A tile: 16x128 bf16, one b128 per thread
  {
    int r  = tid >> 4;
    int c8 = (tid & 15) * 8;
    const uint4* src = (const uint4*)(Qn + (size_t)(bt0 + r) * DIMQ + qc0 + c8);
    *(uint4*)(ldsA + r * LDA_D + c8) = *src;
  }

  // B tile: whole (h,p) key slice (256x128 bf16 = 64KB) via Tensor Data Mover,
  // with TDM LDS-padding producing the 136-element row stride.
  if (w == 0) {
    unsigned lds_base = (unsigned)(uintptr_t)(void*)ldsB;
    unsigned long long ga =
        (unsigned long long)(uintptr_t)(keysbf + (size_t)(h * 2 + p) * NKEYS * DHALF);
    u32x4 g0;
    g0[0] = 1u;                                   // count=1, user mode
    g0[1] = lds_base;                             // lds_addr
    g0[2] = (unsigned)ga;                         // global_addr[31:0]
    g0[3] = (unsigned)((ga >> 32) & 0x1FFFFFFu) | (2u << 30);  // addr[56:32], type=2
    u32x8 g1;
    // data_size=2B, pad_enable, pad_interval=64 dwords (code 5), pad_amount=4 dwords (code 3)
    g1[0] = (1u << 16) | (1u << 20) | (5u << 22) | (3u << 25);
    g1[1] = (32768u & 0xFFFFu) << 16;             // tensor_dim0 = 32768 elems (lo16)
    g1[2] = (32768u >> 16) | (1u << 16);          // tensor_dim0 hi, tensor_dim1 = 1
    g1[3] = (32768u & 0xFFFFu) << 16;             // tile_dim0 = 32768
    g1[4] = 1u;                                   // tile_dim1 = 1
    g1[5] = 32768u;                               // tensor_dim0_stride lo32
    g1[6] = (32768u & 0xFFFFu) << 16;             // stride0 hi, tensor_dim1_stride lo16
    g1[7] = 0u;
    asm volatile("tensor_load_to_lds %0, %1" :: "s"(g0), "s"(g1) : "memory");
  }
  __builtin_amdgcn_s_wait_tensorcnt(0);
  __syncthreads();

  f32x8 acc[2] = {};
#pragma unroll
  for (int dk = 0; dk < DHALF; dk += 32) {
    bf16x16 a = load_A_frag(ldsA + dk, LDA_D, lane);
#pragma unroll
    for (int t = 0; t < 2; ++t) {        // wave w owns key tiles 2w, 2w+1
      bf16x16 b = load_B_frag(ldsB + (w * 2 + t) * 16 * LDA_D + dk, LDA_D, lane);
      acc[t] = wmma_bf16(a, b, acc[t]);
    }
  }
  __syncthreads();
#pragma unroll
  for (int t = 0; t < 2; ++t)
    store_C(scores + (w * 2 + t) * 16, 256, lane, acc[t]);
  __syncthreads();

  // iterative wave-argmax top-16; wave w handles rows w and w+8
  for (int rpass = 0; rpass < 2; ++rpass) {
    int row = w + rpass * 8;
    float v[8];
#pragma unroll
    for (int j = 0; j < 8; ++j) v[j] = scores[row * 256 + lane * 8 + j];
    for (int it = 0; it < TOPKK; ++it) {
      float mv = v[0]; int mn = lane * 8;
#pragma unroll
      for (int j = 1; j < 8; ++j)
        if (v[j] > mv) { mv = v[j]; mn = lane * 8 + j; }
#pragma unroll
      for (int o = 16; o > 0; o >>= 1) {
        float ov = __shfl_xor(mv, o);
        int   on = __shfl_xor(mn, o);
        if (ov > mv || (ov == mv && on < mn)) { mv = ov; mn = on; }
      }
      if ((mn >> 3) == lane) v[mn & 7] = NEG_INF;
      if (lane == 0) {
        size_t o = (((size_t)(bt0 + row) * NHEADS + h) * 2 + p) * TOPKK + it;
        s1ws[o] = mv; i1ws[o] = mn;
      }
    }
  }
}

// ---------------- product-key combine: 256 combo sums -> top-16 -> softmax
__global__ __launch_bounds__(256) void k_combine(const float* __restrict__ s1ws,
                                                 const int* __restrict__ i1ws,
                                                 float* __restrict__ attnws,
                                                 int* __restrict__ vidxws) {
  __shared__ float pv[8][TOPKK];
  __shared__ int   pi[8][TOPKK];
  int tid = threadIdx.x;
  int w = tid >> 5, lane = tid & 31;
  int g  = blockIdx.x * 8 + w;     // flattened (bt, h)
  int bt = g >> 2, h = g & 3;
  size_t base0 = (((size_t)bt * NHEADS + h) * 2 + 0) * TOPKK;
  size_t base1 = (((size_t)bt * NHEADS + h) * 2 + 1) * TOPKK;
  float s0 = 0.f, s1 = 0.f; int i0 = 0, i1 = 0;
  if (lane < 16) {
    s0 = s1ws[base0 + lane]; i0 = i1ws[base0 + lane];
    s1 = s1ws[base1 + lane]; i1 = i1ws[base1 + lane];
  }
  float v[8];
#pragma unroll
  for (int j = 0; j < 8; ++j) {
    int c = lane * 8 + j;
    v[j] = __shfl(s0, c >> 4) + __shfl(s1, c & 15);
  }
  for (int it = 0; it < TOPKK; ++it) {
    float mv = v[0]; int mn = lane * 8;
#pragma unroll
    for (int j = 1; j < 8; ++j)
      if (v[j] > mv) { mv = v[j]; mn = lane * 8 + j; }
#pragma unroll
    for (int o = 16; o > 0; o >>= 1) {
      float ov = __shfl_xor(mv, o);
      int   on = __shfl_xor(mn, o);
      if (ov > mv || (ov == mv && on < mn)) { mv = ov; mn = on; }
    }
    if ((mn >> 3) == lane) v[mn & 7] = NEG_INF;
    int a = mn >> 4, bb = mn & 15;
    int vid = __shfl(i0, a) * NKEYS + __shfl(i1, bb);
    if (lane == 0) { pv[w][it] = mv; pi[w][it] = vid; }
  }
  float m = pv[w][0];                    // picks are descending
  float e = (lane < 16) ? __expf(pv[w][lane] - m) : 0.f;
  float tot = e;
#pragma unroll
  for (int o = 16; o > 0; o >>= 1) tot += __shfl_xor(tot, o);
  if (lane < 16) {
    size_t o = ((size_t)bt * NHEADS + h) * TOPKK + lane;
    attnws[o] = e / tot;
    vidxws[o] = pi[w][lane];
  }
}

// ---------------- weighted value gather (L2-resident values, coalesced rows)
__global__ __launch_bounds__(256) void k_output(const float* __restrict__ attnws,
                                                const int* __restrict__ vidxws,
                                                const float* __restrict__ values,
                                                float* __restrict__ out) {
  __shared__ float a_s[64];
  __shared__ int   v_s[64];
  int tid = threadIdx.x;
  int bt = blockIdx.x;
  if (tid < 64) {
    a_s[tid] = attnws[(size_t)bt * 64 + tid];
    v_s[tid] = vidxws[(size_t)bt * 64 + tid];
  }
  __syncthreads();
  float acc0 = 0.f, acc1 = 0.f;
  for (int k = 0; k < 64; ++k) {
    float wgt = a_s[k];
    const float2* vp = (const float2*)(values + (size_t)v_s[k] * DIMX) + tid;
    if (k + 1 < 64)
      __builtin_prefetch(values + (size_t)v_s[k + 1] * DIMX + tid * 2, 0, 1);
    float2 val = *vp;
    acc0 += wgt * val.x;
    acc1 += wgt * val.y;
  }
  out[(size_t)bt * DIMX + tid * 2]     = acc0;
  out[(size_t)bt * DIMX + tid * 2 + 1] = acc1;
}

extern "C" void kernel_launch(void* const* d_in, const int* in_sizes, int n_in,
                              void* d_out, int out_size, void* d_ws, size_t ws_size,
                              hipStream_t stream) {
  (void)in_sizes; (void)n_in; (void)out_size; (void)ws_size;
  const float* x      = (const float*)d_in[0];
  const float* wq     = (const float*)d_in[1];
  const float* gamma  = (const float*)d_in[2];
  const float* beta   = (const float*)d_in[3];
  const float* keys   = (const float*)d_in[4];
  const float* values = (const float*)d_in[5];
  float* out = (float*)d_out;

  char* ws = (char*)d_ws;
  size_t off = 0;
  float* Q             = (float*)(ws + off);          off += (size_t)BT * DIMQ * 4;
  float* mu            = (float*)(ws + off);          off += DIMQ * 4;
  float* rstd          = (float*)(ws + off);          off += DIMQ * 4;
  unsigned short* Qn   = (unsigned short*)(ws + off); off += (size_t)BT * DIMQ * 2;
  unsigned short* kbf  = (unsigned short*)(ws + off); off += (size_t)NHEADS * 2 * NKEYS * DHALF * 2;
  float* s1ws          = (float*)(ws + off);          off += (size_t)BT * NHEADS * 2 * TOPKK * 4;
  int*   i1ws          = (int*)(ws + off);            off += (size_t)BT * NHEADS * 2 * TOPKK * 4;
  float* attnws        = (float*)(ws + off);          off += (size_t)BT * NHEADS * TOPKK * 4;
  int*   vidxws        = (int*)(ws + off);            off += (size_t)BT * NHEADS * TOPKK * 4;

  k_gemm_q  <<<dim3(BT / 32, DIMQ / 256), 256, 0, stream>>>(x, wq, Q);
  k_bn_stats<<<dim3(DIMQ / 32),            256, 0, stream>>>(Q, mu, rstd);
  k_bn_apply<<<dim3((BT * DIMQ) / 1024),   256, 0, stream>>>(Q, mu, rstd, gamma, beta, Qn);
  k_cvt_keys<<<dim3((NHEADS * 2 * NKEYS * DHALF) / 1024), 256, 0, stream>>>(keys, kbf);
  k_dots_topk<<<dim3(BT / 16, NHEADS * 2), 256, 0, stream>>>(Qn, kbf, s1ws, i1ws);
  k_combine <<<dim3(BT * NHEADS / 8),      256, 0, stream>>>(s1ws, i1ws, attnws, vidxws);
  k_output  <<<dim3(BT),                   256, 0, stream>>>(attnws, vidxws, values, out);
}
